// SpikLinearLayer_24283745092263
// MI455X (gfx1250) — matrix operand order repaired and checked by reading, not compiled, not run on hardware
//
#include <hip/hip_runtime.h>

typedef __attribute__((ext_vector_type(16))) __bf16 v16bf;
typedef __attribute__((ext_vector_type(8)))  __bf16 v8bf;
typedef __attribute__((ext_vector_type(8)))  float  v8f;
typedef __attribute__((ext_vector_type(4)))  unsigned int v4u;
typedef __attribute__((ext_vector_type(8)))  int v8i_;
typedef __attribute__((ext_vector_type(4)))  int v4i_;

#define T_STEPS 16
#define BATCH   1024
#define DIN     1024
#define DOUT    1024
#define ROWS    (T_STEPS * BATCH)   // 16384

// GEMM tiling
#define TM 128
#define TN 128
#define TK 32
#define K_ITERS (DIN / TK)          // 32
#define LDS_ROW_BYTES 80            // 64 data bytes + 16 pad (TDM pad_enable)
#define TILE_BYTES (128 * LDS_ROW_BYTES)

// ---------------------------------------------------------------------------
// helpers
// ---------------------------------------------------------------------------
__device__ __forceinline__ unsigned lds_off(const void* p) {
  // generic LDS pointer = {SHARED_BASE aperture, 32-bit LDS byte offset}
  return (unsigned)(unsigned long long)p;
}

__device__ __forceinline__ unsigned short f32_bf16(float f) {
  unsigned u = __builtin_bit_cast(unsigned, f);
  unsigned r = u + 0x7FFFu + ((u >> 16) & 1u);   // round-to-nearest-even
  return (unsigned short)(r >> 16);
}

// TDM: load a (128 rows x TK cols) bf16 tile of a row-major [tensor_rows x row_len]
// tensor into LDS, padding each 64B row to 80B (pad_interval=16 DW, pad_amount=4 DW).
__device__ __forceinline__ void tdm_load_tile(unsigned lds_addr,
                                              unsigned long long gaddr,
                                              unsigned tensor_rows,
                                              unsigned row_len) {
  v4u g0;
  g0.x = 1u;                                        // count=1, user mode
  g0.y = lds_addr;                                  // LDS byte address
  g0.z = (unsigned)gaddr;                           // global_addr[31:0]
  g0.w = ((unsigned)(gaddr >> 32) & 0x01FFFFFFu)    // global_addr[56:32]
         | (2u << 30);                              // type = 2 ("image")
  v8i_ g1;
  g1[0] = (int)((1u << 16)        // data_size = 1 -> 2 bytes
              | (1u << 20)        // pad_enable
              | (3u << 22)        // pad_interval code 3 -> 16 DWORDs (64B)
              | (3u << 25));      // pad_amount  code 3 -> 4 DWORDs (16B)
  g1[1] = (int)((row_len & 0xFFFFu) << 16);                       // tensor_dim0 lo
  g1[2] = (int)((row_len >> 16) | ((tensor_rows & 0xFFFFu) << 16)); // dim0 hi | dim1 lo
  g1[3] = (int)((tensor_rows >> 16) | ((unsigned)TK << 16));        // dim1 hi | tile_dim0
  g1[4] = (int)128u;              // tile_dim1 = 128 rows, tile_dim2 = 0
  g1[5] = (int)row_len;           // tensor_dim0_stride[31:0] (elements)
  g1[6] = 0;                      // stride0 hi | tensor_dim1_stride lo (unused, 2D)
  g1[7] = 0;
  v4i_ gz = {0, 0, 0, 0};
#if __clang_major__ >= 23
  v8i_ gz8 = {0, 0, 0, 0, 0, 0, 0, 0};
  __builtin_amdgcn_tensor_load_to_lds(g0, g1, gz, gz, gz8, 0);
#else
  __builtin_amdgcn_tensor_load_to_lds(g0, g1, gz, gz, 0);
#endif
}

// ---------------------------------------------------------------------------
// 1) f32 -> bf16 conversion (x and W), float4-vectorized
// ---------------------------------------------------------------------------
__global__ __launch_bounds__(256)
void cvt_f32_bf16(const float* __restrict__ in, unsigned short* __restrict__ out,
                  int n4) {
  int i = blockIdx.x * blockDim.x + threadIdx.x;
  if (i >= n4) return;
  float4 v = ((const float4*)in)[i];
  ushort4 o;
  o.x = f32_bf16(v.x); o.y = f32_bf16(v.y);
  o.z = f32_bf16(v.z); o.w = f32_bf16(v.w);
  ((ushort4*)out)[i] = o;
}

// ---------------------------------------------------------------------------
// 2) bf16 WMMA GEMM: h[ROWS][DOUT] = xb[ROWS][DIN] * wb[DOUT][DIN]^T
//    128x128 block tile, 8 waves (4 M x 2 N), 32x64 per wave (2x4 WMMA tiles),
//    TDM double-buffered K slices of 32.
// ---------------------------------------------------------------------------
__global__ __launch_bounds__(256)
void snn_gemm_bf16(const unsigned short* __restrict__ xb,
                   const unsigned short* __restrict__ wb,
                   float* __restrict__ h) {
  __shared__ __align__(128) unsigned char sA[2][TILE_BYTES];
  __shared__ __align__(128) unsigned char sB[2][TILE_BYTES];

  const int tid    = threadIdx.x;
  const int wid    = tid >> 5;
  const int lane   = tid & 31;
  const int half   = lane >> 4;   // 0: lanes 0-15, 1: lanes 16-31
  const int l16    = lane & 15;
  const int wave_m = wid >> 1;    // 0..3
  const int wave_n = wid & 1;     // 0..1

  const int m0 = blockIdx.y * TM;
  const int n0 = blockIdx.x * TN;

  const unsigned long long a_base =
      (unsigned long long)xb + (unsigned long long)m0 * DIN * 2ull;
  const unsigned long long b_base =
      (unsigned long long)wb + (unsigned long long)n0 * DIN * 2ull;

  v8f acc[2][4] = {};

  if (wid == 0) tdm_load_tile(lds_off(&sA[0][0]), a_base, ROWS, DIN);
  if (wid == 1) tdm_load_tile(lds_off(&sB[0][0]), b_base, DOUT, DIN);
  if (wid < 2) __builtin_amdgcn_s_wait_tensorcnt(0);
  __syncthreads();

  for (int kk = 0; kk < K_ITERS; ++kk) {
    const int cur = kk & 1;
    if (kk + 1 < K_ITERS) {  // prefetch next K slice into the other buffer
      const unsigned long long koff = (unsigned long long)(kk + 1) * TK * 2ull;
      if (wid == 0) tdm_load_tile(lds_off(&sA[cur ^ 1][0]), a_base + koff, ROWS, DIN);
      if (wid == 1) tdm_load_tile(lds_off(&sB[cur ^ 1][0]), b_base + koff, DOUT, DIN);
    }

    const unsigned char* Ab = &sA[cur][0];
    const unsigned char* Bb = &sB[cur][0];

    // A fragments: 16x32 bf16. Lane half 0: K {0..7,16..23}; half 1: {8..15,24..31}
    v16bf afrag[2];
#pragma unroll
    for (int mt = 0; mt < 2; ++mt) {
      const unsigned char* rp =
          Ab + (unsigned)(wave_m * 32 + mt * 16 + l16) * LDS_ROW_BYTES;
      v8bf lo = *(const v8bf*)(rp + half * 16);
      v8bf hi = *(const v8bf*)(rp + 32 + half * 16);
      afrag[mt] = __builtin_shufflevector(lo, hi, 0, 1, 2, 3, 4, 5, 6, 7,
                                          8, 9, 10, 11, 12, 13, 14, 15);
    }
    // B fragments: 32x16 bf16; B[k][n] = W[n][k] -> lane holds K of one W row:
    // half 0 -> K {0..7,16..23}, half 1 -> K {8..15,24..31}.
    v16bf bfrag[4];
#pragma unroll
    for (int nt = 0; nt < 4; ++nt) {
      const unsigned char* rp =
          Bb + (unsigned)(wave_n * 64 + nt * 16 + l16) * LDS_ROW_BYTES + half * 16;
      v8bf lo = *(const v8bf*)(rp);
      v8bf hi = *(const v8bf*)(rp + 32);
      bfrag[nt] = __builtin_shufflevector(lo, hi, 0, 1, 2, 3, 4, 5, 6, 7,
                                          8, 9, 10, 11, 12, 13, 14, 15);
    }

#pragma unroll
    for (int mt = 0; mt < 2; ++mt)
#pragma unroll
      for (int nt = 0; nt < 4; ++nt)
        acc[mt][nt] = __builtin_amdgcn_wmma_f32_16x16x32_bf16(
            false, afrag[mt], false, bfrag[nt], (short)0, acc[mt][nt],
            false, false);

    if (kk + 1 < K_ITERS) {
      if (wid < 2) __builtin_amdgcn_s_wait_tensorcnt(0);
      __syncthreads();
    }
  }

  // C/D layout: VGPR r -> M = half*8 + r, N = l16 (per 16x16 f32 table)
#pragma unroll
  for (int mt = 0; mt < 2; ++mt) {
#pragma unroll
    for (int nt = 0; nt < 4; ++nt) {
      const int gm = m0 + wave_m * 32 + mt * 16 + half * 8;
      const int gn = n0 + wave_n * 64 + nt * 16 + l16;
      float* dst = h + (long long)gm * DOUT + gn;
#pragma unroll
      for (int r = 0; r < 8; ++r)
        dst[(long long)r * DOUT] = acc[mt][nt][r];
    }
  }
}

// ---------------------------------------------------------------------------
// 3) BN stats: per-column mean/var over 16384 rows -> fused scale/bias
// ---------------------------------------------------------------------------
__global__ __launch_bounds__(256)
void bn_stats(const float* __restrict__ h, const float* __restrict__ gamma,
              const float* __restrict__ beta, float* __restrict__ scale,
              float* __restrict__ bias) {
  __shared__ float red[2][256];
  const int tid = threadIdx.x;
  const int c = blockIdx.x * 64 + (tid & 63);
  const int r0 = tid >> 6;
  float s = 0.f, s2 = 0.f;
  for (int r = r0; r < ROWS; r += 4) {
    float v = h[(long long)r * DOUT + c];
    s += v;
    s2 += v * v;
  }
  red[0][tid] = s;
  red[1][tid] = s2;
  __syncthreads();
  if (tid < 64) {
    s  = red[0][tid] + red[0][tid + 64] + red[0][tid + 128] + red[0][tid + 192];
    s2 = red[1][tid] + red[1][tid + 64] + red[1][tid + 128] + red[1][tid + 192];
    const float inv_n = 1.0f / (float)ROWS;
    const float mean = s * inv_n;
    const float var  = s2 * inv_n - mean * mean;   // biased, like jnp.var
    const float sc = gamma[c] * rsqrtf(var + 1e-5f);
    scale[c] = sc;
    bias[c]  = beta[c] - mean * sc;
  }
}

// ---------------------------------------------------------------------------
// 4) LIF dynamics over T=16: v += (x-v)/2, spike = (v>=1), hard reset
// ---------------------------------------------------------------------------
__global__ __launch_bounds__(256)
void lif_kernel(const float* __restrict__ h, const float* __restrict__ scale,
                const float* __restrict__ bias, float* __restrict__ out) {
  const int idx = blockIdx.x * 256 + threadIdx.x;  // (b, d) flattened
  const int d = idx & (DOUT - 1);
  const float sc = scale[d];
  const float bi = bias[d];
  float v = 0.f;
#pragma unroll
  for (int t = 0; t < T_STEPS; ++t) {
    const float xt = h[(long long)t * (BATCH * DOUT) + idx] * sc + bi;
    v += (xt - v) * 0.5f;
    const float spk = (v >= 1.0f) ? 1.0f : 0.0f;
    out[(long long)t * (BATCH * DOUT) + idx] = spk;
    v = (1.0f - spk) * v;
  }
}

// ---------------------------------------------------------------------------
extern "C" void kernel_launch(void* const* d_in, const int* in_sizes, int n_in,
                              void* d_out, int out_size, void* d_ws, size_t ws_size,
                              hipStream_t stream) {
  const float* x     = (const float*)d_in[0];  // [16,1024,1024]
  const float* W     = (const float*)d_in[1];  // [1024,1024]
  const float* gamma = (const float*)d_in[2];  // [1024]
  const float* beta  = (const float*)d_in[3];  // [1024]
  float* out = (float*)d_out;                  // [16,1024,1024]

  char* ws = (char*)d_ws;
  float* h = (float*)ws;                                           // 64 MiB
  unsigned short* xb = (unsigned short*)(ws + (size_t)ROWS * DOUT * 4);   // 32 MiB
  unsigned short* wb = (unsigned short*)((char*)xb + (size_t)ROWS * DIN * 2); // 2 MiB
  float* scale = (float*)((char*)wb + (size_t)DOUT * DIN * 2);
  float* bias  = scale + DOUT;

  const int nx4 = ROWS * DIN / 4;
  const int nw4 = DOUT * DIN / 4;
  cvt_f32_bf16<<<(nx4 + 255) / 256, 256, 0, stream>>>(x, xb, nx4);
  cvt_f32_bf16<<<(nw4 + 255) / 256, 256, 0, stream>>>(W, wb, nw4);

  snn_gemm_bf16<<<dim3(DOUT / TN, ROWS / TM), 256, 0, stream>>>(xb, wb, h);

  bn_stats<<<DOUT / 64, 256, 0, stream>>>(h, gamma, beta, scale, bias);

  lif_kernel<<<(BATCH * DOUT) / 256, 256, 0, stream>>>(h, scale, bias, out);
}
